// StridedAttention_1254130450884
// MI455X (gfx1250) — matrix operand order repaired and checked by reading, not compiled, not run on hardware
//
#include <hip/hip_runtime.h>

// ---------------------------------------------------------------------------
// Strided (block-causal) attention, MI455X / gfx1250, bf16 WMMA path.
// B=2, T=2048, D=2048, H=16, HD=128, STRIDE=64.
// Round 3: double-buffered LDS staging with GLOBAL_LOAD_ASYNC_TO_LDS_B128
// (ASYNCcnt-tracked), plus global_prefetch of far tiles.
// ---------------------------------------------------------------------------

typedef __attribute__((ext_vector_type(8)))  __bf16 v8bf;
typedef __attribute__((ext_vector_type(16))) __bf16 v16bf;
typedef __attribute__((ext_vector_type(8)))  float  v8f;

#define NB 2
#define NT 2048
#define ND 2048
#define NH 16
#define HD 128
#define MTOT (NB * NT)   // 4096 GEMM rows

#define HAS_ASYNC_LDS __has_builtin(__builtin_amdgcn_global_load_async_to_lds_b128)

// Builtin expects int4-vector pointers: addrspace(1) source, addrspace(3) dest.
typedef __attribute__((__vector_size__(4 * sizeof(int)))) int i4vec;
typedef __attribute__((address_space(1))) i4vec as1_i4;
typedef __attribute__((address_space(3))) i4vec as3_i4;

// 16-byte global -> LDS copy. Async (ASYNCcnt-tracked, no VGPR round trip)
// when available; otherwise a plain b128 load + ds_store_b128.
__device__ __forceinline__ void cp16_g2l(const unsigned short* g,
                                         unsigned short* l) {
#if HAS_ASYNC_LDS
  __builtin_amdgcn_global_load_async_to_lds_b128(
      (as1_i4*)(unsigned long long)(g),
      (as3_i4*)(unsigned)(unsigned long long)(l), 0, 0);
#else
  *(uint4*)l = *(const uint4*)g;
#endif
}

__device__ __forceinline__ void wait_async_zero() {
#if HAS_ASYNC_LDS
#if __has_builtin(__builtin_amdgcn_s_wait_asynccnt)
  __builtin_amdgcn_s_wait_asynccnt(0);
#else
  asm volatile("s_wait_asynccnt 0" ::: "memory");
#endif
#endif
}

// fp32 -> bf16 round-to-nearest-even, bit-level (no header dependence)
__device__ __forceinline__ unsigned short f2bf(float f) {
  unsigned int u = __builtin_bit_cast(unsigned int, f);
  u += 0x7FFFu + ((u >> 16) & 1u);
  return (unsigned short)(u >> 16);
}

// Build a 16-element bf16 fragment from two contiguous 8-element halves.
__device__ __forceinline__ v16bf load_frag16(const unsigned short* p, int hi_off) {
  v8bf lo = *(const v8bf*)(p);
  v8bf hi = *(const v8bf*)(p + hi_off);
  return __builtin_shufflevector(lo, hi, 0,1,2,3,4,5,6,7,8,9,10,11,12,13,14,15);
}

__device__ __forceinline__ v8f wmma_bf16(v16bf a, v16bf b, v8f c) {
  return __builtin_amdgcn_wmma_f32_16x16x32_bf16(
      /*neg_a=*/false, a, /*neg_b=*/false, b,
      /*c_mod=*/(short)0, c, /*reuse_a=*/false, /*reuse_b=*/false);
}

// ---------------------------------------------------------------------------
// Kernel 1: fp32 -> bf16 conversion (grid-stride)
// ---------------------------------------------------------------------------
__global__ void cvt_f32_bf16(const float* __restrict__ src,
                             unsigned short* __restrict__ dst, int n) {
  for (int i = blockIdx.x * blockDim.x + threadIdx.x; i < n;
       i += gridDim.x * blockDim.x)
    dst[i] = f2bf(src[i]);
}

// ---------------------------------------------------------------------------
// Kernel 2/4: GEMM  Y[m,n] = sum_k A[m,k] * W[n,k]   (y = x @ W.T)
// MODE 0: write bf16 with [B,H,T,HD] permute (for q/k/v)
// MODE 1: write fp32 row-major [m,n] (final output)
// 128x128 tile, 256 threads (8 waves), K-step 64, double-buffered LDS.
// Wave w: rows (w&3)*32..+31, cols (w>>2)*64..+63  -> 2x4 accum tiles.
// ---------------------------------------------------------------------------
template <int MODE>
__global__ __launch_bounds__(256) void gemm_bf16_nt(
    const unsigned short* __restrict__ A, const unsigned short* __restrict__ W,
    unsigned short* __restrict__ dst16, float* __restrict__ dstf) {
  __shared__ unsigned short sA[2][128 * 64];  // [m][k]
  __shared__ unsigned short sB[2][128 * 64];  // [n][k] (== B[k][n] of matmul)

  const int tid  = threadIdx.x;
  const int lane = tid & 31;
  const int wave = tid >> 5;
  const int rw = wave & 3;   // row group (32 rows)
  const int cw = wave >> 2;  // col group (64 cols)
  const int ln = lane & 15;
  const int half = (lane >= 16) ? 1 : 0;

  const int m0 = blockIdx.y * 128;
  const int n0 = blockIdx.x * 128;

  v8f acc[2][4];
#pragma unroll
  for (int i = 0; i < 2; ++i)
#pragma unroll
    for (int j = 0; j < 4; ++j) acc[i][j] = (v8f){0, 0, 0, 0, 0, 0, 0, 0};

  // Stage one K-tile (128x64 of A and of W) into LDS buffer `b`.
  auto stage = [&](int b, int kk) {
#pragma unroll
    for (int c = 0; c < 4; ++c) {  // 1024 16-byte chunks / 256 threads
      int chunk = tid + c * 256;
      int row = chunk >> 3;
      int col = (chunk & 7) << 3;
      cp16_g2l(&A[(size_t)(m0 + row) * ND + kk + col], &sA[b][row * 64 + col]);
      cp16_g2l(&W[(size_t)(n0 + row) * ND + kk + col], &sB[b][row * 64 + col]);
    }
  };

  stage(0, 0);
  int buf = 0;
  for (int kk = 0; kk < ND; kk += 64) {
    wait_async_zero();   // this wave's staged data is in LDS
    __syncthreads();     // everyone's buf data visible; buf^1 readers done
    if (kk + 64 < ND) {
      // pull the tile after next into cache ahead of the async engine
      if (kk + 128 < ND) {
        int r = tid & 127;
        const unsigned short* p = (tid < 128)
                                      ? &A[(size_t)(m0 + r) * ND + kk + 128]
                                      : &W[(size_t)(n0 + r) * ND + kk + 128];
        __builtin_prefetch(p, 0, 0);
      }
      stage(buf ^ 1, kk + 64);
    }

#pragma unroll
    for (int ks = 0; ks < 64; ks += 32) {
      v16bf fa[2], fb[4];
      const int kbA = ks + half * 8;   // A: K {0..7,16..23} / {8..15,24..31}
      const int kbB = ks + half * 8;   // B: K {0..7,16..23} / {8..15,24..31}
#pragma unroll
      for (int i = 0; i < 2; ++i)
        fa[i] = load_frag16(&sA[buf][(rw * 32 + i * 16 + ln) * 64 + kbA], 16);
#pragma unroll
      for (int j = 0; j < 4; ++j)
        fb[j] = load_frag16(&sB[buf][(cw * 64 + j * 16 + ln) * 64 + kbB], 16);
#pragma unroll
      for (int i = 0; i < 2; ++i)
#pragma unroll
        for (int j = 0; j < 4; ++j)
          acc[i][j] = wmma_bf16(fa[i], fb[j], acc[i][j]);
    }
    buf ^= 1;
  }

  // Epilogue. C layout: VGPR r, lane l -> row (l>=16?8:0)+r, col l&15.
#pragma unroll
  for (int i = 0; i < 2; ++i)
#pragma unroll
    for (int j = 0; j < 4; ++j)
#pragma unroll
      for (int r = 0; r < 8; ++r) {
        int m = m0 + rw * 32 + i * 16 + half * 8 + r;
        int n = n0 + cw * 64 + j * 16 + ln;
        float val = acc[i][j][r];
        if (MODE == 0) {
          int b = m >> 11, t = m & (NT - 1);
          int h = n >> 7, d = n & (HD - 1);
          dst16[((size_t)(b * NH + h) * NT + t) * HD + d] = f2bf(val);
        } else {
          dstf[(size_t)m * ND + n] = val;
        }
      }
}

// ---------------------------------------------------------------------------
// Kernel 3: flash attention with block-causal mask (granularity 64 == tiles).
// Grid: B*H*(T/64) blocks, 128 threads (4 waves). Wave w owns query rows
// t0+16w .. t0+16w+15, full HD=128. Key blocks j = 0..qb (no per-element
// masking: STRIDE == key-tile == query-tile = 64). Double-buffered K/V tiles.
// ---------------------------------------------------------------------------
__global__ __launch_bounds__(128) void attn_kernel(
    const unsigned short* __restrict__ Q, const unsigned short* __restrict__ K,
    const unsigned short* __restrict__ V, unsigned short* __restrict__ O) {
  __shared__ unsigned short sK[2][64 * 128];   // [key][d]
  __shared__ unsigned short sVT[2][128 * 64];  // [d][key] (V transposed)
  __shared__ unsigned short sP[4 * 16 * 64];   // per-wave P scratch [m][key]

  const int tid  = threadIdx.x;
  const int lane = tid & 31;
  const int wave = tid >> 5;
  const int ln = lane & 15;
  const int half = (lane >= 16) ? 1 : 0;

  const int bid = blockIdx.x;
  const int qb = bid & 31;          // T/64 = 32 query blocks
  const int h  = (bid >> 5) & 15;
  const int b  = bid >> 9;
  const int t0 = qb * 64;

  const size_t headoff = (size_t)(b * NH + h) * NT * HD;
  const float scale = 0.08838834764831845f;  // 1/sqrt(128)

  // Q fragments for this wave's 16 rows (4 k-chunks of 32 over HD=128)
  const unsigned short* qrowp = Q + headoff + (size_t)(t0 + wave * 16 + ln) * HD;
  v16bf fq[4];
#pragma unroll
  for (int kc = 0; kc < 4; ++kc)
    fq[kc] = load_frag16(qrowp + kc * 32 + half * 8, 16);

  v8f o[8];
#pragma unroll
  for (int nt = 0; nt < 8; ++nt) o[nt] = (v8f){0, 0, 0, 0, 0, 0, 0, 0};
  float mrow[8], lrow[8];
#pragma unroll
  for (int r = 0; r < 8; ++r) { mrow[r] = -1e30f; lrow[r] = 0.0f; }

  // Stage key block j into buffer `bsel`: K async/b128, V transposed (scatter).
  auto stage_kv = [&](int bsel, int j) {
    const unsigned short* kp = K + headoff + (size_t)j * 64 * HD;
    const unsigned short* vp = V + headoff + (size_t)j * 64 * HD;
#pragma unroll
    for (int c = 0; c < 8; ++c) {  // 8192 elems / 128 thr = 8 x b128
      int chunk = tid + c * 128;
      int row = chunk >> 4;
      int col = (chunk & 15) << 3;
      cp16_g2l(&kp[row * 128 + col], &sK[bsel][row * 128 + col]);
    }
#pragma unroll
    for (int c = 0; c < 8; ++c) {
      int chunk = tid + c * 128;
      int row = chunk >> 4;         // key
      int col = (chunk & 15) << 3;  // d base
      uint4 t4 = *(const uint4*)&vp[row * 128 + col];
      unsigned short e[8];
      *(uint4*)e = t4;
#pragma unroll
      for (int q2 = 0; q2 < 8; ++q2) sVT[bsel][(col + q2) * 64 + row] = e[q2];
    }
  };

  stage_kv(0, 0);
  int buf = 0;
  for (int j = 0; j <= qb; ++j) {
    wait_async_zero();
    __syncthreads();
    if (j < qb) stage_kv(buf ^ 1, j + 1);

    // S = Q K^T * scale : 4 tiles of 16x16 (keys), K-dim = HD in 4 steps.
    v8f s[4];
#pragma unroll
    for (int kt = 0; kt < 4; ++kt) {
      s[kt] = (v8f){0, 0, 0, 0, 0, 0, 0, 0};
#pragma unroll
      for (int kc = 0; kc < 4; ++kc) {
        v16bf fk = load_frag16(
            &sK[buf][(kt * 16 + ln) * 128 + kc * 32 + half * 8], 16);
        s[kt] = wmma_bf16(fq[kc], fk, s[kt]);
      }
    }

    // Online softmax. Row m = half*8 + r lives in VGPR r, cols across the 16
    // lanes of this half-wave -> xor-shuffle reductions (masks 1..8 in-half).
    unsigned short* pw = &sP[wave * 16 * 64];
#pragma unroll
    for (int r = 0; r < 8; ++r) {
      float mx = -1e30f;
#pragma unroll
      for (int kt = 0; kt < 4; ++kt) {
        s[kt][r] *= scale;
        mx = fmaxf(mx, s[kt][r]);
      }
#pragma unroll
      for (int off = 1; off < 16; off <<= 1)
        mx = fmaxf(mx, __shfl_xor(mx, off, 32));
      float nm = fmaxf(mrow[r], mx);
      float cs = __expf(mrow[r] - nm);
      float sum = 0.0f;
#pragma unroll
      for (int kt = 0; kt < 4; ++kt) {
        float p = __expf(s[kt][r] - nm);
        s[kt][r] = p;
        sum += p;
      }
#pragma unroll
      for (int off = 1; off < 16; off <<= 1) sum += __shfl_xor(sum, off, 32);
      lrow[r] = lrow[r] * cs + sum;
      mrow[r] = nm;
#pragma unroll
      for (int nt = 0; nt < 8; ++nt) o[nt][r] *= cs;  // rescale accumulator
#pragma unroll
      for (int kt = 0; kt < 4; ++kt)  // spill P [m][key] to wave-private LDS
        pw[(half * 8 + r) * 64 + kt * 16 + ln] = f2bf(s[kt][r]);
    }
    // Fence P stores before re-reading as A fragments (same-wave DS).
    asm volatile("s_wait_dscnt 0" ::: "memory");

    v16bf fp[2];
#pragma unroll
    for (int kc2 = 0; kc2 < 2; ++kc2)
      fp[kc2] = load_frag16(&pw[ln * 64 + kc2 * 32 + half * 8], 16);

    // O += P V : 8 d-tiles, K-dim = 64 keys in 2 steps.
#pragma unroll
    for (int nt = 0; nt < 8; ++nt)
#pragma unroll
      for (int kc2 = 0; kc2 < 2; ++kc2) {
        v16bf fv = load_frag16(
            &sVT[buf][(nt * 16 + ln) * 64 + kc2 * 32 + half * 8], 16);
        o[nt] = wmma_bf16(fp[kc2], fv, o[nt]);
      }
    buf ^= 1;
  }

  // Normalize and write att in [B,T,D] (= [B,T,H*HD]) bf16 for the final GEMM.
#pragma unroll
  for (int r = 0; r < 8; ++r) {
    float inv = 1.0f / lrow[r];
    size_t row = (size_t)(b * NT + t0 + wave * 16 + half * 8 + r);
#pragma unroll
    for (int nt = 0; nt < 8; ++nt)
      O[row * ND + h * HD + nt * 16 + ln] = f2bf(o[nt][r] * inv);
  }
}

// ---------------------------------------------------------------------------
// Launch
// ---------------------------------------------------------------------------
extern "C" void kernel_launch(void* const* d_in, const int* in_sizes, int n_in,
                              void* d_out, int out_size, void* d_ws,
                              size_t ws_size, hipStream_t stream) {
  const float* x  = (const float*)d_in[0];
  const float* wq = (const float*)d_in[1];
  const float* wk = (const float*)d_in[2];
  const float* wv = (const float*)d_in[3];
  const float* wo = (const float*)d_in[4];

  const size_t EX = (size_t)MTOT * ND;  // 8,388,608 elems
  const size_t EW = (size_t)ND * ND;    // 4,194,304 elems

  unsigned short* ws = (unsigned short*)d_ws;
  unsigned short* xb  = ws;
  unsigned short* wqb = xb + EX;
  unsigned short* wkb = wqb + EW;
  unsigned short* wvb = wkb + EW;
  unsigned short* wob = wvb + EW;
  unsigned short* qb  = wob + EW;
  unsigned short* kb  = qb + EX;
  unsigned short* vb  = kb + EX;
  unsigned short* att = vb + EX;

  cvt_f32_bf16<<<4096, 256, 0, stream>>>(x, xb, (int)EX);
  cvt_f32_bf16<<<2048, 256, 0, stream>>>(wq, wqb, (int)EW);
  cvt_f32_bf16<<<2048, 256, 0, stream>>>(wk, wkb, (int)EW);
  cvt_f32_bf16<<<2048, 256, 0, stream>>>(wv, wvb, (int)EW);
  cvt_f32_bf16<<<2048, 256, 0, stream>>>(wo, wob, (int)EW);

  dim3 gg(ND / 128, MTOT / 128);  // (16, 32)
  gemm_bf16_nt<0><<<gg, 256, 0, stream>>>(xb, wqb, qb, nullptr);
  gemm_bf16_nt<0><<<gg, 256, 0, stream>>>(xb, wkb, kb, nullptr);
  gemm_bf16_nt<0><<<gg, 256, 0, stream>>>(xb, wvb, vb, nullptr);

  attn_kernel<<<NB * NH * (NT / 64), 128, 0, stream>>>(qb, kb, vb, att);

  gemm_bf16_nt<1><<<gg, 256, 0, stream>>>(att, wob, nullptr, (float*)d_out);
}
